// InputPreprocessor_42374147342445
// MI455X (gfx1250) — matrix operand order repaired and checked by reading, not compiled, not run on hardware
//
#include <hip/hip_runtime.h>
#include <hip/hip_bf16.h>

typedef __attribute__((ext_vector_type(16))) _Float16 v16h;
typedef __attribute__((ext_vector_type(8)))  float    v8f;

#define N_EL  32
#define N_ION 8
#define NF    32
#define OUTC  65

// ---- A-matrix K index for slot s (8 packed f16 pairs per lane), per ISA 16-bit A layout ----
__device__ __forceinline__ int kA_map(int s, int half) {
    int k = (s < 4) ? (2 * s) : (16 + 2 * (s - 4));
    return k + 8 * half;
}

__device__ __forceinline__ v8f zero8() {
    v8f z;
#pragma unroll
    for (int i = 0; i < 8; ++i) z[i] = 0.0f;
    return z;
}

__device__ __forceinline__ v8f wmma16(v16h a, v16h b, v8f c) {
    // D = A(16x32 f16) x B(32x16 f16) + C(16x16 f32)
    return __builtin_amdgcn_wmma_f32_16x16x32_f16(false, a, false, b, (short)0, c, false, false);
}

// Branchless tanh: 1 - 2/(exp(2x)+1).  exp->inf => +1, exp->0 => -1.
__device__ __forceinline__ float tanh_fast(float x) {
    float e = __expf(2.0f * x);
    return 1.0f - 2.0f * __builtin_amdgcn_rcpf(e + 1.0f);
}

// Per-lane radial-basis constants for this lane's 16 K-slots (loop-invariant).
struct RbfParams { float mu[16]; float rs[16]; };

__device__ __forceinline__ void init_rbf(RbfParams& p, int half) {
#pragma unroll
    for (int s = 0; s < 8; ++s) {
        int k = kA_map(s, half);
#pragma unroll
        for (int j = 0; j < 2; ++j) {
            float q   = (float)(k + j) * (1.0f / 31.0f);
            float sig = (1.0f + 5.0f * q) * (1.0f / 7.0f);
            p.mu[2 * s + j] = q * q * 5.0f;
            p.rs[2 * s + j] = __builtin_amdgcn_rcpf(sig);
        }
    }
}

// Load one 32x16 B tile (row-major [K][32] weights) into B layout:
// lane n holds column n%16, K in the same slot order as the A layout (kA_map).
__device__ __forceinline__ v16h load_Btile(const float* __restrict__ W, int col,
                                           int row_base, int nrows, int half) {
    v16h bt;
#pragma unroll
    for (int s = 0; s < 8; ++s) {
        int k = row_base + kA_map(s, half);
        float v0 = (k     < nrows) ? W[(k    ) * NF + col] : 0.0f;
        float v1 = (k + 1 < nrows) ? W[(k + 1) * NF + col] : 0.0f;
        bt[2 * s    ] = (_Float16)v0;
        bt[2 * s + 1] = (_Float16)v1;
    }
    return bt;
}

// Layer-1 A tiles in-register: K 0..31 = rbf features, K 32..63 = [dist, dx, dy, dz, 0...]
__device__ __forceinline__ void build_A(const RbfParams& p, float d, float dx, float dy, float dz,
                                        int half, v16h& alo, v16h& ahi) {
    float d2 = d * d;
#pragma unroll
    for (int s = 0; s < 8; ++s) {
#pragma unroll
        for (int j = 0; j < 2; ++j) {
            float t = (d - p.mu[2 * s + j]) * p.rs[2 * s + j];
            alo[2 * s + j] = (_Float16)(d2 * __expf(-d - t * t));
        }
        int kp = 32 + kA_map(s, half);
        float v0 = 0.0f, v1 = 0.0f;
        if (kp == 32) { v0 = d;  v1 = dx; }
        else if (kp == 34) { v0 = dy; v1 = dz; }
        ahi[2 * s    ] = (_Float16)v0;
        ahi[2 * s + 1] = (_Float16)v1;
    }
}

struct Net {
    v16h B1lo0, B1lo1, B1hi0, B1hi1, B20, B21, Bg0, Bg1;
    float b1_0, b1_1, b2_0, b2_1;
};

__device__ __forceinline__ void load_net(Net& nt, const float* W1, const float* b1,
                                         const float* W2, const float* b2,
                                         const float* Wg, int lane, int w1rows) {
    int half = lane >> 4, mrow = lane & 15;
    int col0 = mrow, col1 = 16 + mrow;
    nt.B1lo0 = load_Btile(W1, col0,  0, w1rows, half);
    nt.B1lo1 = load_Btile(W1, col1,  0, w1rows, half);
    nt.B1hi0 = load_Btile(W1, col0, 32, w1rows, half);
    nt.B1hi1 = load_Btile(W1, col1, 32, w1rows, half);
    nt.B20   = load_Btile(W2, col0,  0, 32, half);
    nt.B21   = load_Btile(W2, col1,  0, 32, half);
    nt.Bg0   = load_Btile(Wg, col0,  0, 32, half);
    nt.Bg1   = load_Btile(Wg, col1,  0, 32, half);
    nt.b1_0 = b1[col0]; nt.b1_1 = b1[col1];
    nt.b2_0 = b2[col0]; nt.b2_1 = b2[col1];
}

// One 16-row tile through MLP + gate; per-lane neighbor-partial sums for columns
// col0 (= lane&15) and col1 (= 16 + lane&15), summed over the lane's 8 C rows.
__device__ __forceinline__ void tile_forward(const Net& nt, v16h Alo, v16h Ahi,
                                             _Float16* __restrict__ h1, int lane,
                                             float& s0, float& s1) {
    int half = lane >> 4, mrow = lane & 15;
    int col0 = mrow, col1 = 16 + mrow;

    // ---- layer 1 (K padded 36 -> 64), tanh, stage to LDS ----
    {
        v8f c = zero8();
        c = wmma16(Alo, nt.B1lo0, c);
        c = wmma16(Ahi, nt.B1hi0, c);
#pragma unroll
        for (int v = 0; v < 8; ++v)
            h1[(v + 8 * half) * NF + col0] = (_Float16)tanh_fast(c[v] + nt.b1_0);
    }
    {
        v8f c = zero8();
        c = wmma16(Alo, nt.B1lo1, c);
        c = wmma16(Ahi, nt.B1hi1, c);
#pragma unroll
        for (int v = 0; v < 8; ++v)
            h1[(v + 8 * half) * NF + col1] = (_Float16)tanh_fast(c[v] + nt.b1_1);
    }

    // ---- reload tanh(h1) as A tile (wave-private LDS; compiler inserts dscnt waits) ----
    v16h A2;
#pragma unroll
    for (int s = 0; s < 8; ++s) {
        int k = kA_map(s, half);
        A2[2 * s    ] = h1[mrow * NF + k];
        A2[2 * s + 1] = h1[mrow * NF + k + 1];
    }

    // ---- layer 2 + gate (gate reuses Alo: rbf32 == first 32 features) ----
    {
        v8f c2 = zero8(); c2 = wmma16(A2,  nt.B20, c2);
        v8f g  = zero8(); g  = wmma16(Alo, nt.Bg0, g);
        float acc = 0.0f;
#pragma unroll
        for (int v = 0; v < 8; ++v) acc += tanh_fast(c2[v] + nt.b2_0) * g[v];
        s0 = acc;
    }
    {
        v8f c2 = zero8(); c2 = wmma16(A2,  nt.B21, c2);
        v8f g  = zero8(); g  = wmma16(Alo, nt.Bg1, g);
        float acc = 0.0f;
#pragma unroll
        for (int v = 0; v < 8; ++v) acc += tanh_fast(c2[v] + nt.b2_1) * g[v];
        s1 = acc;
    }
}

__global__ __launch_bounds__(256)
void deeperwin_edges_wmma(const float* __restrict__ r, const float* __restrict__ Rion,
                          const float* __restrict__ Wee1, const float* __restrict__ bee1,
                          const float* __restrict__ Wee2, const float* __restrict__ bee2,
                          const float* __restrict__ Wee_g,
                          const float* __restrict__ Wei1, const float* __restrict__ bei1,
                          const float* __restrict__ Wei2, const float* __restrict__ bei2,
                          const float* __restrict__ Wei_g,
                          const int* __restrict__ n_up_p,
                          float* __restrict__ out) {
    __shared__ float     accE[N_EL * NF];      // 4 KB: h_ee accumulator
    __shared__ _Float16  h1s[8][16 * NF];      // 8 KB: per-wave tanh(h1) staging

    const int b    = blockIdx.x;
    const int tid  = threadIdx.x;
    const int lane = tid & 31;
    const int wave = tid >> 5;
    const int half = lane >> 4;
    const int mrow = lane & 15;

    for (int i = tid; i < N_EL * NF; i += 256) accE[i] = 0.0f;
    __syncthreads();

    RbfParams rp;
    init_rbf(rp, half);

    const float* rb   = r + (size_t)b * (N_EL * 3);
    _Float16*    h1   = &h1s[wave][0];
    float*       outb = out + (size_t)b * (N_EL * OUTC);

    // ================= el-el: 64 row tiles of 16 pairs =================
    {
        Net nt;
        load_net(nt, Wee1, bee1, Wee2, bee2, Wee_g, lane, 36);
        for (int t = wave; t < (N_EL * N_EL) / 16; t += 8) {
            int pair = t * 16 + mrow;
            int ie = pair >> 5, je = pair & 31;
            float dx = rb[ie * 3 + 0] - rb[je * 3 + 0];
            float dy = rb[ie * 3 + 1] - rb[je * 3 + 1];
            float dz = rb[ie * 3 + 2] - rb[je * 3 + 2];
            float d  = sqrtf(dx * dx + dy * dy + dz * dz + 1e-12f);
            v16h Alo, Ahi;
            build_A(rp, d, dx, dy, dz, half, Alo, Ahi);
            float s0, s1;
            tile_forward(nt, Alo, Ahi, h1, lane, s0, s1);
            // whole tile shares electron i = t>>1; merge 2 tiles x 2 lane-halves via LDS atomics
            int ieT = t >> 1;
            atomicAdd(&accE[ieT * NF +      mrow], s0);
            atomicAdd(&accE[ieT * NF + 16 + mrow], s1);
        }
    }

    // ================= el-ion: 16 row tiles (2 electrons x 8 ions each) =================
    {
        Net nt;
        load_net(nt, Wei1, bei1, Wei2, bei2, Wei_g, lane, 36);
        for (int t = wave; t < (N_EL * N_ION) / 16; t += 8) {
            int pair = t * 16 + mrow;
            int ie = pair >> 3, io = pair & 7;
            float dx = rb[ie * 3 + 0] - Rion[io * 3 + 0];
            float dy = rb[ie * 3 + 1] - Rion[io * 3 + 1];
            float dz = rb[ie * 3 + 2] - Rion[io * 3 + 2];
            float d  = sqrtf(dx * dx + dy * dy + dz * dz + 1e-12f);
            v16h Alo, Ahi;
            build_A(rp, d, dx, dy, dz, half, Alo, Ahi);
            float s0, s1;
            tile_forward(nt, Alo, Ahi, h1, lane, s0, s1);
            // lane's 8 C rows == all 8 ions of electron e = 2t + half -> full ion sum, direct store
            int e = 2 * t + half;
            outb[e * OUTC + 32 + mrow] = s0;
            outb[e * OUTC + 48 + mrow] = s1;
        }
    }

    __syncthreads();

    // ================= emit h_ee and spin =================
    for (int i = tid; i < N_EL * NF; i += 256) {
        int ie = i >> 5, nf = i & 31;
        outb[ie * OUTC + nf] = accE[i];
    }
    if (tid < N_EL) {
        int nup = *n_up_p;
        outb[tid * OUTC + 64] = (tid < nup) ? 1.0f : -1.0f;
    }
}

extern "C" void kernel_launch(void* const* d_in, const int* in_sizes, int n_in,
                              void* d_out, int out_size, void* d_ws, size_t ws_size,
                              hipStream_t stream) {
    const int*   n_up = (const int*)d_in[0];
    const float* r    = (const float*)d_in[2];
    const float* R    = (const float*)d_in[3];
    const float* Wee1 = (const float*)d_in[5];
    const float* bee1 = (const float*)d_in[6];
    const float* Wee2 = (const float*)d_in[7];
    const float* bee2 = (const float*)d_in[8];
    const float* Weeg = (const float*)d_in[9];
    const float* Wei1 = (const float*)d_in[10];
    const float* bei1 = (const float*)d_in[11];
    const float* Wei2 = (const float*)d_in[12];
    const float* bei2 = (const float*)d_in[13];
    const float* Weig = (const float*)d_in[14];
    float* out = (float*)d_out;

    const int B = in_sizes[2] / (N_EL * 3);   // 1024
    deeperwin_edges_wmma<<<dim3(B), dim3(256), 0, stream>>>(
        r, R, Wee1, bee1, Wee2, bee2, Weeg, Wei1, bei1, Wei2, bei2, Weig, n_up, out);
}
